// DownSampleGlobal_43267500540732
// MI455X (gfx1250) — compile-verified, matrix operand order checked by reading, not run on hardware
//
#include <hip/hip_runtime.h>
#include <hip/hip_bf16.h>

// ---------------------------------------------------------------------------
// DownSampleGlobal on MI455X (gfx1250): f16 WMMA pipeline, contiguous fragments
//   B=8, C=128, N=2048, H=4, d=32, M=1024
// All contraction dims are stored pi-permuted (pi swaps halves [8..15]<->[16..23]
// within each 32-block) so BOTH WMMA A and B fragments are single 32B loads.
// ---------------------------------------------------------------------------

typedef _Float16 f16;
typedef __attribute__((ext_vector_type(16))) _Float16 v16h;
typedef __attribute__((ext_vector_type(8)))  _Float16 v8h;
typedef __attribute__((ext_vector_type(8)))  float    v8f;

#define B_DIM 8
#define C_DIM 128
#define N_DIM 2048
#define H_DIM 4
#define D_DIM 32
#define M_KEEP 1024
#define SM_SCALE 0.17677669529663687f  // 1/sqrt(32)

// pi within a 32-block: [0-7]->[0-7], [8-15]->[16-23], [16-23]->[8-15], [24-31] id.
// Involution. Maps the gapped WMMA-A half pattern onto contiguous [g*16, g*16+15].
__device__ inline int kperm(int kk) {
  int grp = (kk >> 3) & 3;
  return (grp == 1) ? kk + 8 : (grp == 2) ? kk - 8 : kk;
}

// Unified WMMA fragment loader on pi-permuted storage (works for A and B):
//   lane: rr = lane&15 selects the row (A: m-row, B: n-col), g = lane>>4.
//   One contiguous 32B v16h load: rowptr + k0 + g*16.
__device__ inline v16h load_frag(const f16* __restrict__ src, int ld, int r0, int k0) {
  int lane = threadIdx.x & 31, g = lane >> 4, rr = lane & 15;
  return *(const v16h*)(src + (size_t)(r0 + rr) * ld + k0 + g * 16);
}

// ---------------------------------------------------------------------------
// Kernel 1: f32 -> f16, transpose x to (B,N,C) with pi-permuted C, permute W
// channel dims, zero col-sum accumulator.
// ---------------------------------------------------------------------------
__global__ void convert_kernel(const float* __restrict__ x,
                               const float* __restrict__ wq,
                               const float* __restrict__ wk,
                               const float* __restrict__ wv,
                               f16* __restrict__ xT, f16* __restrict__ whq,
                               f16* __restrict__ whk, f16* __restrict__ whv,
                               float* __restrict__ colsum) {
  size_t i = (size_t)blockIdx.x * blockDim.x + threadIdx.x;
  const size_t totalX = (size_t)B_DIM * C_DIM * N_DIM;  // 2,097,152
  if (i < totalX) {
    int n = (int)(i & (N_DIM - 1));
    int c = (int)((i >> 11) & (C_DIM - 1));
    int b = (int)(i >> 18);
    int cs = (c & ~31) | kperm(c & 31);
    xT[((size_t)b * N_DIM + n) * C_DIM + cs] = (f16)x[i];
  }
  if (i < (size_t)C_DIM * C_DIM) {
    int c = (int)(i & 127), o = (int)(i >> 7);
    size_t d = (size_t)o * C_DIM + ((c & ~31) | kperm(c & 31));
    whq[d] = (f16)wq[i]; whk[d] = (f16)wk[i]; whv[d] = (f16)wv[i];
  }
  if (i < (size_t)B_DIM * H_DIM * N_DIM) colsum[i] = 0.0f;
}

// ---------------------------------------------------------------------------
// Kernel 2: QKV projection, no LDS. qT,kT: (B,N,128) pi-perm head dims.
//   vH: (B,128,N) pi-perm sequence dim. 8 waves x 3 tile jobs x 4 K-steps.
// ---------------------------------------------------------------------------
__global__ __launch_bounds__(256) void proj_kernel(const f16* __restrict__ xT,
                                                   const f16* __restrict__ whq,
                                                   const f16* __restrict__ whk,
                                                   const f16* __restrict__ whv,
                                                   f16* __restrict__ qT,
                                                   f16* __restrict__ kT,
                                                   f16* __restrict__ vH) {
  int b  = blockIdx.x >> 7;
  int n0 = (blockIdx.x & 127) * 16;
  int t = threadIdx.x, wave = t >> 5, lane = t & 31, g = lane >> 4, colx = lane & 15;
  const f16* xb = xT + (size_t)b * N_DIM * C_DIM;

  #pragma unroll
  for (int i = 0; i < 3; ++i) {
    int job = wave * 3 + i;      // 0..23
    int kind = job >> 3;         // 0:qT 1:kT 2:v
    int o0 = (job & 7) * 16;
    v8f acc = {};
    if (kind < 2) {
      // D[n][o] = sum_c xT[n][c] W[o][c]   (contraction over pi-perm c)
      const f16* W = (kind == 0) ? whq : whk;
      #pragma unroll
      for (int ks = 0; ks < 4; ++ks) {
        v16h a  = load_frag(xb, C_DIM, n0, ks * 32);
        v16h bf = load_frag(W,  C_DIM, o0, ks * 32);
        acc = __builtin_amdgcn_wmma_f32_16x16x32_f16(false, a, false, bf,
                                                     (short)0, acc, false, false);
      }
      // store with head-dim pi permutation (o0&16 == within-block base)
      f16* dst = ((kind == 0) ? qT : kT) + ((size_t)b * N_DIM + n0) * C_DIM;
      int slot = (o0 & ~31) + kperm((o0 & 16) + colx);
      #pragma unroll
      for (int r = 0; r < 8; ++r)
        dst[(size_t)(r + g * 8) * C_DIM + slot] = (f16)acc[r];
    } else {
      // D[o][n] = sum_c Wv[o][c] x[c][n]
      v8f acc2 = {};
      #pragma unroll
      for (int ks = 0; ks < 4; ++ks) {
        v16h a  = load_frag(whv, C_DIM, o0, ks * 32);
        v16h bf = load_frag(xb,  C_DIM, n0, ks * 32);
        acc2 = __builtin_amdgcn_wmma_f32_16x16x32_f16(false, a, false, bf,
                                                      (short)0, acc2, false, false);
      }
      // store with sequence-dim pi permutation
      f16* dst = vH + ((size_t)b * C_DIM + o0) * N_DIM;
      int slot = (n0 & ~31) + kperm((n0 & 16) + colx);
      #pragma unroll
      for (int r = 0; r < 8; ++r)
        dst[(size_t)(r + g * 8) * N_DIM + slot] = (f16)acc2[r];
    }
  }
}

// ---------------------------------------------------------------------------
// Kernel 3: attention strip, one WG = (b, h, 16 rows).
//   phase1: 128 WMMAs -> scaled energy strip (16x2048 f16 LDS, pi-perm cols)
//   phase2: row max + exp (UNNORMALIZED, inv deferred), vectorized v8h LDS
//   phase2b: col sums (inv folded per row), vectorized + 8 atomics/thread
//   phase3: 128 WMMAs attn@v^T (pi-consistent with vH), LDS reduce, scale by inv
// ---------------------------------------------------------------------------
__global__ __launch_bounds__(256) void attn_kernel(const f16* __restrict__ qT,
                                                   const f16* __restrict__ kT,
                                                   const f16* __restrict__ vH,
                                                   float* __restrict__ O,
                                                   float* __restrict__ colsum) {
  __shared__ f16   sE[16 * N_DIM];   // 64 KB
  __shared__ float sPart[8 * 256];   // 8 KB
  __shared__ float sInv[16];

  int blk = blockIdx.x;
  int b  = blk >> 9;
  int h  = (blk >> 7) & 3;
  int n0 = (blk & 127) * 16;
  int t = threadIdx.x, wave = t >> 5, lane = t & 31, g = lane >> 4, colx = lane & 15;

  // ---- phase 1: energy ---------------------------------------------------
  const f16* qb = qT + (size_t)b * N_DIM * C_DIM;
  const f16* kb = kT + (size_t)b * N_DIM * C_DIM;
  v16h aq = load_frag(qb, C_DIM, n0, h * D_DIM);
  for (int mt = wave; mt < 128; mt += 8) {
    v16h bk = load_frag(kb, C_DIM, mt * 16, h * D_DIM);
    v8f e = {};
    e = __builtin_amdgcn_wmma_f32_16x16x32_f16(false, aq, false, bk,
                                               (short)0, e, false, false);
    int slot = ((mt >> 1) << 5) + kperm(((mt & 1) << 4) + colx);
    #pragma unroll
    for (int r = 0; r < 8; ++r)
      sE[(r + g * 8) * N_DIM + slot] = (f16)(e[r] * SM_SCALE);
  }
  __syncthreads();

  // ---- phase 2: row max + unnormalized exp (16 threads/row, v8h chunks) --
  {
    int row = t >> 4, cid = t & 15;
    f16* rp = sE + row * N_DIM + cid * 128;
    float mx = -3.0e38f;
    #pragma unroll
    for (int ch = 0; ch < 16; ++ch) {
      v8h v = *(const v8h*)(rp + ch * 8);
      #pragma unroll
      for (int j = 0; j < 8; ++j) mx = fmaxf(mx, (float)v[j]);
    }
    #pragma unroll
    for (int m = 1; m < 16; m <<= 1) mx = fmaxf(mx, __shfl_xor(mx, m, 32));
    float s = 0.0f;
    #pragma unroll
    for (int ch = 0; ch < 16; ++ch) {
      v8h v = *(const v8h*)(rp + ch * 8);
      v8h e;
      #pragma unroll
      for (int j = 0; j < 8; ++j) {
        float ev = __expf((float)v[j] - mx);
        s += ev;
        e[j] = (f16)ev;
      }
      *(v8h*)(rp + ch * 8) = e;
    }
    #pragma unroll
    for (int m = 1; m < 16; m <<= 1) s += __shfl_xor(s, m, 32);
    if (cid == 0) sInv[row] = 1.0f / s;
  }
  __syncthreads();

  // ---- phase 2b: column sums (un-permute on write) ------------------------
  {
    int base = t * 8;                                   // pi-slot base (8-aligned)
    int orig = (base & ~31) + kperm(base & 31);         // true column base
    float a8[8] = {0, 0, 0, 0, 0, 0, 0, 0};
    #pragma unroll
    for (int r = 0; r < 16; ++r) {
      float iv = sInv[r];
      v8h v = *(const v8h*)(sE + r * N_DIM + base);
      #pragma unroll
      for (int j = 0; j < 8; ++j) a8[j] += iv * (float)v[j];
    }
    float* cb = colsum + (size_t)(b * H_DIM + h) * N_DIM + orig;
    #pragma unroll
    for (int j = 0; j < 8; ++j) atomicAdd(cb + j, a8[j]);
  }

  // ---- phase 3: O = attn @ v^T (2 dd-tiles x 4 K-ranges across 8 waves) ---
  {
    int tt = wave >> 2, kr = wave & 3;
    const f16* vb = vH + ((size_t)b * C_DIM + h * D_DIM + tt * 16) * N_DIM;
    v8f acc = {};
    for (int kk = 0; kk < 16; ++kk) {
      int k0 = kr * 512 + kk * 32;
      v16h a  = load_frag(sE, N_DIM, 0, k0);   // pi-slot K, matches vH storage
      v16h bb = load_frag(vb, N_DIM, 0, k0);
      acc = __builtin_amdgcn_wmma_f32_16x16x32_f16(false, a, false, bb,
                                                   (short)0, acc, false, false);
    }
    #pragma unroll
    for (int r = 0; r < 8; ++r)
      sPart[wave * 256 + (r + g * 8) * 16 + colx] = acc[r];
  }
  __syncthreads();

  // ---- reduce partials, apply deferred softmax norm, write O (B,H,N,32) ---
  {
    int row = t >> 4, c = t & 15;
    float iv = sInv[row];
    #pragma unroll
    for (int tt = 0; tt < 2; ++tt) {
      float s = 0.0f;
      #pragma unroll
      for (int kr = 0; kr < 4; ++kr) s += sPart[(tt * 4 + kr) * 256 + t];
      O[((size_t)(b * H_DIM + h) * N_DIM + n0 + row) * D_DIM + tt * 16 + c] = s * iv;
    }
  }
}

// ---------------------------------------------------------------------------
// Kernel 4: per-(b,h) bitonic sort (descending, ascending-index tie-break),
// then gather. d_out (float): [ x_ds | idx | x_dropped | idx_dropped ]
// ---------------------------------------------------------------------------
__global__ __launch_bounds__(256) void sort_gather_kernel(const float* __restrict__ colsum,
                                                          const float* __restrict__ O,
                                                          float* __restrict__ out) {
  __shared__ float key[N_DIM];
  __shared__ int   sidx[N_DIM];
  int bh = blockIdx.x, b = bh >> 2, h = bh & 3;
  int t = threadIdx.x;

  const float* c = colsum + (size_t)bh * N_DIM;
  for (int i = t; i < N_DIM; i += 256) { key[i] = c[i]; sidx[i] = i; }
  __syncthreads();

  for (int k = 2; k <= N_DIM; k <<= 1) {
    for (int j = k >> 1; j > 0; j >>= 1) {
      for (int i = t; i < N_DIM; i += 256) {
        int ij = i ^ j;
        if (ij > i) {
          float ka = key[i], kb = key[ij];
          int   ia = sidx[i], ib = sidx[ij];
          bool before = (ka > kb) || (ka == kb && ia < ib);  // descending
          if (before == ((i & k) != 0)) {
            key[i] = kb; key[ij] = ka; sidx[i] = ib; sidx[ij] = ia;
          }
        }
      }
      __syncthreads();
    }
  }

  const size_t XDS  = 0;
  const size_t IDX  = (size_t)B_DIM * C_DIM * M_KEEP;              // 1,048,576
  const size_t XDR  = IDX + (size_t)B_DIM * H_DIM * M_KEEP;        // +32,768
  const size_t IDXD = XDR + (size_t)B_DIM * C_DIM * M_KEEP;

  for (int m = t; m < M_KEEP; m += 256) {
    out[IDX  + (size_t)bh * M_KEEP + m] = (float)sidx[m];
    out[IDXD + (size_t)bh * M_KEEP + m] = (float)sidx[N_DIM - 1 - m];
  }

  const float* Obh = O + (size_t)bh * N_DIM * D_DIM;
  for (int e = t; e < M_KEEP * D_DIM; e += 256) {
    int m = e >> 5, dd = e & 31;
    size_t dst = ((size_t)b * C_DIM + h * D_DIM + dd) * M_KEEP + m;
    out[XDS + dst] = Obh[(size_t)sidx[m] * D_DIM + dd];
    out[XDR + dst] = Obh[(size_t)sidx[N_DIM - 1 - m] * D_DIM + dd];
  }
}

// ---------------------------------------------------------------------------
extern "C" void kernel_launch(void* const* d_in, const int* in_sizes, int n_in,
                              void* d_out, int out_size, void* d_ws, size_t ws_size,
                              hipStream_t stream) {
  const float* x  = (const float*)d_in[0];
  const float* Wq = (const float*)d_in[1];
  const float* Wk = (const float*)d_in[2];
  const float* Wv = (const float*)d_in[3];
  float* out = (float*)d_out;

  char* ws = (char*)d_ws;
  const size_t XT_B = (size_t)B_DIM * N_DIM * C_DIM * 2;           // 4 MB
  const size_t W_B  = (size_t)C_DIM * C_DIM * 2;                   // 32 KB each
  const size_t QT_B = (size_t)B_DIM * N_DIM * C_DIM * 2;           // 4 MB
  const size_t O_B  = (size_t)B_DIM * H_DIM * N_DIM * D_DIM * 4;   // 8 MB
  f16*   xT  = (f16*)(ws);
  f16*   whq = (f16*)(ws + XT_B);
  f16*   whk = (f16*)(ws + XT_B + W_B);
  f16*   whv = (f16*)(ws + XT_B + 2 * W_B);
  f16*   qT  = (f16*)(ws + XT_B + 3 * W_B);
  f16*   kT  = (f16*)(ws + XT_B + 3 * W_B + QT_B);
  f16*   vH  = (f16*)(ws + XT_B + 3 * W_B + 2 * QT_B);
  float* O   = (float*)(ws + XT_B + 3 * W_B + 3 * QT_B);
  float* col = (float*)(ws + XT_B + 3 * W_B + 3 * QT_B + O_B);

  const size_t totalX = (size_t)B_DIM * C_DIM * N_DIM;
  convert_kernel<<<(unsigned)((totalX + 255) / 256), 256, 0, stream>>>(
      x, Wq, Wk, Wv, xT, whq, whk, whv, col);

  proj_kernel<<<B_DIM * (N_DIM / 16), 256, 0, stream>>>(xT, whq, whk, whv, qT, kT, vH);

  attn_kernel<<<B_DIM * H_DIM * (N_DIM / 16), 256, 0, stream>>>(qT, kT, vH, O, col);

  sort_gather_kernel<<<B_DIM * H_DIM, 256, 0, stream>>>(col, O, out);
}
